// QuantumTSTransformer_16544214024235
// MI455X (gfx1250) — compile-verified, matrix operand order checked by reading, not run on hardware
//
#include <hip/hip_runtime.h>
#include <math.h>

// ---------------------------------------------------------------------------
// Quantum TS transformer, fully fused: one workgroup per batch element.
// - State (2^14 complex amps = 128KB) lives entirely in LDS (CDNA5 WGP: 320KB).
// - Angle projection GEMM done in-block with v_wmma_f32_16x16x32_f16.
// - All ancilla-only circuit sections (prepare / prepare-adjoint / pcphase)
//   are pre-contracted into five 16x16 unitaries applied as single LDS passes.
// - Commuting RY blocks of sim14 are pair-fused (two qubits per pass).
// ---------------------------------------------------------------------------

namespace {

constexpr int kNQ   = 10;
constexpr int kNW   = 14;
constexpr int kT    = 16;
constexpr int kF    = 64;
constexpr int kNROT = 80;            // 4*NQ*LAYERS
constexpr int kQFF  = 40;            // 4*NQ*1
constexpr int kDW   = 1 << kNW;      // 16384 amplitudes
constexpr int kPairs  = kDW / 2;     // 8192
constexpr int kCPairs = kDW / 4;     // 4096
constexpr int kThr  = 1024;

typedef __attribute__((ext_vector_type(16))) _Float16 v16h;
typedef __attribute__((ext_vector_type(8)))  float    v8f;

__device__ __forceinline__ int strideOf(int q) { return 1 << (kNW - 1 - q); }

// Element j (0..15) of a v16h A/B fragment for V_WMMA_F32_16X16X32_F16,
// per ISA 7.12.2 (16-bit A 16x32 layout; B symmetric with M<->N).
__device__ __forceinline__ int kmap16(int j, int half) {
  int v = j >> 1, e = j & 1;
  return (v < 4) ? (half * 8 + 2 * v + e) : (16 + half * 8 + 2 * (v - 4) + e);
}

// ---- gate primitives on the LDS-resident state ----------------------------

// controlled RX: [[c,-is],[-is,c]] on target where control==1, scalar angle
__device__ __forceinline__ void g_crx(float* re, float* im, int cst, int tst,
                                      float c, float s, int tid) {
  const int hi = cst > tst ? cst : tst;
  const int lo = cst > tst ? tst : cst;
  const int lm = lo - 1, hm = hi - 1;
  for (int p = tid; p < kCPairs; p += kThr) {
    int t0 = ((p & ~lm) << 1) | (p & lm);
    int i  = ((t0 & ~hm) << 1) | (t0 & hm);
    int i0 = i | cst;
    int i1 = i0 | tst;
    float r0 = re[i0], m0 = im[i0], r1 = re[i1], m1 = im[i1];
    re[i0] = c * r0 + s * m1;  im[i0] = c * m0 - s * r1;
    re[i1] = c * r1 + s * m0;  im[i1] = c * m1 - s * r0;
  }
}

// fused RY(qubit A) * RY(qubit B), scalar angles (commuting; real coeffs)
__device__ __forceinline__ void g_ryry(float* re, float* im, int sa, int sb,
                                       float ca, float sA, float cb, float sB,
                                       int tid) {
  const int hi = sa > sb ? sa : sb;
  const int lo = sa > sb ? sb : sa;
  const int lm = lo - 1, hm = hi - 1;
  for (int p = tid; p < kCPairs; p += kThr) {
    int t0  = ((p & ~lm) << 1) | (p & lm);
    int i00 = ((t0 & ~hm) << 1) | (t0 & hm);
    int i10 = i00 | sa, i01 = i00 | sb, i11 = i10 | sb;
    float r00 = re[i00], m00 = im[i00], r01 = re[i01], m01 = im[i01];
    float r10 = re[i10], m10 = im[i10], r11 = re[i11], m11 = im[i11];
    float t00r = ca * r00 - sA * r10, t00i = ca * m00 - sA * m10;
    float t10r = sA * r00 + ca * r10, t10i = sA * m00 + ca * m10;
    float t01r = ca * r01 - sA * r11, t01i = ca * m01 - sA * m11;
    float t11r = sA * r01 + ca * r11, t11i = sA * m01 + ca * m11;
    re[i00] = cb * t00r - sB * t01r;  im[i00] = cb * t00i - sB * t01i;
    re[i01] = sB * t00r + cb * t01r;  im[i01] = sB * t00i + cb * t01i;
    re[i10] = cb * t10r - sB * t11r;  im[i10] = cb * t10i - sB * t11i;
    re[i11] = sB * t10r + cb * t11r;  im[i11] = sB * t10i + cb * t11i;
  }
}

// select variants: angle depends on t = low 4 bits (ancilla index); all select
// strides are >=16 so (index & 15) is invariant within a pair/quadruple.
__device__ __forceinline__ void g_ryry_sel(float* re, float* im, int sa, int sb,
                                           const float* csA, const float* snA,
                                           const float* csB, const float* snB,
                                           float sgn, int tid) {
  const int hi = sa > sb ? sa : sb;
  const int lo = sa > sb ? sb : sa;
  const int lm = lo - 1, hm = hi - 1;
  for (int p = tid; p < kCPairs; p += kThr) {
    int t0  = ((p & ~lm) << 1) | (p & lm);
    int i00 = ((t0 & ~hm) << 1) | (t0 & hm);
    int i10 = i00 | sa, i01 = i00 | sb, i11 = i10 | sb;
    int t = i00 & 15;
    float ca = csA[t], sA = sgn * snA[t];
    float cb = csB[t], sB = sgn * snB[t];
    float r00 = re[i00], m00 = im[i00], r01 = re[i01], m01 = im[i01];
    float r10 = re[i10], m10 = im[i10], r11 = re[i11], m11 = im[i11];
    float t00r = ca * r00 - sA * r10, t00i = ca * m00 - sA * m10;
    float t10r = sA * r00 + ca * r10, t10i = sA * m00 + ca * m10;
    float t01r = ca * r01 - sA * r11, t01i = ca * m01 - sA * m11;
    float t11r = sA * r01 + ca * r11, t11i = sA * m01 + ca * m11;
    re[i00] = cb * t00r - sB * t01r;  im[i00] = cb * t00i - sB * t01i;
    re[i01] = sB * t00r + cb * t01r;  im[i01] = sB * t00i + cb * t01i;
    re[i10] = cb * t10r - sB * t11r;  im[i10] = cb * t10i - sB * t11i;
    re[i11] = sB * t10r + cb * t11r;  im[i11] = sB * t10i + cb * t11i;
  }
}

__device__ __forceinline__ void g_crx_sel(float* re, float* im, int cst, int tst,
                                          const float* cs, const float* sn,
                                          float sgn, int tid) {
  const int hi = cst > tst ? cst : tst;
  const int lo = cst > tst ? tst : cst;
  const int lm = lo - 1, hm = hi - 1;
  for (int p = tid; p < kCPairs; p += kThr) {
    int t0 = ((p & ~lm) << 1) | (p & lm);
    int i  = ((t0 & ~hm) << 1) | (t0 & hm);
    int i0 = i | cst;
    int i1 = i0 | tst;
    int t = i0 & 15;
    float c = cs[t], s = sgn * sn[t];
    float r0 = re[i0], m0 = im[i0], r1 = re[i1], m1 = im[i1];
    re[i0] = c * r0 + s * m1;  im[i0] = c * m0 - s * r1;
    re[i1] = c * r1 + s * m0;  im[i1] = c * m1 - s * r0;
  }
}

// apply 16x16 ancilla-space unitary M to every contiguous 16-amp block.
// M stored transposed: Mre[t*16 + j] = M[j][t]. 16 lanes cooperate per block
// (broadcast state reads, conflict-free matrix reads, contiguous writes).
__device__ __forceinline__ void applyA(float* re, float* im,
                                       const float* Mre, const float* Mim,
                                       int tid) {
  const int wave = tid >> 5, lane = tid & 31;
  const int sub = lane >> 4;
  const int j = lane & 15;
  for (int it = 0; it < 16; ++it) {
    int base = (it * 64 + wave * 2 + sub) * 16;
    float ar = 0.0f, ai = 0.0f;
#pragma unroll
    for (int t = 0; t < 16; ++t) {
      float xr = re[base + t], xi = im[base + t];
      float mr = Mre[t * 16 + j], mi = Mim[t * 16 + j];
      ar += mr * xr - mi * xi;
      ai += mr * xi + mi * xr;
    }
    // wave lockstep: all 16 reads above retire before these writes issue
    re[base + j] = ar;
    im[base + j] = ai;
  }
}

// ---- sim14 segment drivers ------------------------------------------------

__device__ __forceinline__ void sel_ry_block(float* re, float* im,
                                             const float* csel, const float* ssel,
                                             int gstart, float sgn, int tid) {
  for (int pr = 0; pr < 5; ++pr) {            // gates commute -> order free
    int g0 = gstart + 2 * pr;
    g_ryry_sel(re, im, strideOf(2 * pr), strideOf(2 * pr + 1),
               &csel[g0 * kT], &ssel[g0 * kT],
               &csel[(g0 + 1) * kT], &ssel[(g0 + 1) * kT], sgn, tid);
    __syncthreads();
  }
}

__device__ __forceinline__ void sel_crx_block1(float* re, float* im,
                                               const float* csel, const float* ssel,
                                               int base, bool adj, float sgn, int tid) {
  for (int step = 0; step < 10; ++step) {     // gates: k=0..9, i=9-k, (i, i+1)
    int k = adj ? 9 - step : step;
    int g = base + 10 + k;
    int i = 9 - k;
    g_crx_sel(re, im, strideOf(i), strideOf((i + 1) % 10),
              &csel[g * kT], &ssel[g * kT], sgn, tid);
    __syncthreads();
  }
}

__device__ __forceinline__ void sel_crx_block2(float* re, float* im,
                                               const float* csel, const float* ssel,
                                               int base, bool adj, float sgn, int tid) {
  for (int step = 0; step < 10; ++step) {     // gates: k=0..9, i=(k?k-1:9), (i, i-1)
    int k = adj ? 9 - step : step;
    int g = base + 30 + k;
    int i = (k == 0) ? 9 : (k - 1);
    g_crx_sel(re, im, strideOf(i), strideOf((i + 9) % 10),
              &csel[g * kT], &ssel[g * kT], sgn, tid);
    __syncthreads();
  }
}

__device__ __forceinline__ void run_select(float* re, float* im,
                                           const float* csel, const float* ssel,
                                           bool adj, int tid) {
  const float sgn = adj ? -1.0f : 1.0f;
  if (!adj) {
    for (int L = 0; L < 2; ++L) {
      int base = L * 40;
      sel_ry_block (re, im, csel, ssel, base +  0, sgn, tid);
      sel_crx_block1(re, im, csel, ssel, base, false, sgn, tid);
      sel_ry_block (re, im, csel, ssel, base + 20, sgn, tid);
      sel_crx_block2(re, im, csel, ssel, base, false, sgn, tid);
    }
  } else {
    for (int L = 1; L >= 0; --L) {
      int base = L * 40;
      sel_crx_block2(re, im, csel, ssel, base, true, sgn, tid);
      sel_ry_block (re, im, csel, ssel, base + 20, sgn, tid);
      sel_crx_block1(re, im, csel, ssel, base, true, sgn, tid);
      sel_ry_block (re, im, csel, ssel, base +  0, sgn, tid);
    }
  }
}

__global__ __launch_bounds__(kThr, 1)
void qts_fused_kernel(const float* __restrict__ x,    // (B, 64, 16)
                      const float* __restrict__ Wp,   // (80, 64)
                      const float* __restrict__ bp,   // (80,)
                      const float* __restrict__ prep, // (4,4,2)
                      const float* __restrict__ sig,  // (5,)
                      const float* __restrict__ qff,  // (40,)
                      const float* __restrict__ Wout, // (1,30)
                      const float* __restrict__ bout, // (1,)
                      float* __restrict__ out) {
  __shared__ float sre[kDW];                   // 64 KB
  __shared__ float sim_[kDW];                  // 64 KB
  // 12 KB pool: phase A = f16 WMMA staging; phase B = ancilla matrices
  __shared__ __align__(16) float poolF[3072];
  __shared__ float csel[kNROT * kT];           // 5 KB  cos(theta/2) per (rot,t)
  __shared__ float ssel[kNROT * kT];           // 5 KB  sin(theta/2)
  __shared__ float pcp[32], psp[32];           // prepare half-angle trig
  __shared__ float qcp[kQFF], qsp[kQFF];       // qff half-angle trig
  __shared__ float sgc[5], sgs[5];             // signal phases
  __shared__ float wred[32];                   // cross-wave reduction scratch

  _Float16* h16  = reinterpret_cast<_Float16*>(poolF);          // 1024 halves
  _Float16* wp16 = reinterpret_cast<_Float16*>(poolF) + 1024;   // 5120 halves
  float* Ure = poolF;                          // 256  (after WMMA phase)
  float* Uim = poolF + 256;                    // 256
  float* Are = poolF + 512;                    // 5*256
  float* Aim = poolF + 1792;                   // 5*256

  const int tid  = threadIdx.x;
  const int b    = blockIdx.x;
  const int wave = tid >> 5;
  const int lane = tid & 31;

  // ---- stage h = x^T + positional encoding (f16), W_proj (f16) ----
  {
    int t = tid >> 6, f = tid & 63;            // 16*64 == 1024 threads
    int e = f & ~1;
    float d  = __expf((float)e * (-9.210340371976184f / 64.0f));
    float pe = (f & 1) ? __cosf((float)t * d) : __sinf((float)t * d);
    float hv = x[(size_t)b * (kF * kT) + f * kT + t] + pe;
    h16[t * kF + f] = (_Float16)hv;
  }
  for (int i = tid; i < kNROT * kF; i += kThr) wp16[i] = (_Float16)Wp[i];
  if (tid < 32) {
    float a = 0.5f * prep[tid];
    pcp[tid] = __cosf(a); psp[tid] = __sinf(a);
  } else if (tid < 32 + kQFF) {
    int k = tid - 32; float a = 0.5f * qff[k];
    qcp[k] = __cosf(a); qsp[k] = __sinf(a);
  } else if (tid < 32 + kQFF + 5) {
    int k = tid - 32 - kQFF; float a = sig[k];
    sgc[k] = __cosf(a); sgs[k] = __sinf(a);
  }
  for (int i = tid; i < kDW; i += kThr) {      // |0...0>
    sre[i] = (i == 0) ? 1.0f : 0.0f;
    sim_[i] = 0.0f;
  }
  __syncthreads();

  // ---- angle projection via WMMA: (16 x 64) @ (64 x 80) in 5 N-tiles ----
  if (wave < 5) {
    const int half  = lane >> 4;
    const int mn    = lane & 15;               // A row / B col within tile
    const int rbase = wave * 16;
    v8f c = {};
    for (int k0 = 0; k0 < kF; k0 += 32) {
      v16h a, bb;
#pragma unroll
      for (int j = 0; j < 16; ++j) {
        int kk = kmap16(j, half);
        a[j]  = h16[mn * kF + k0 + kk];
        bb[j] = wp16[(rbase + mn) * kF + k0 + kk];
      }
      c = __builtin_amdgcn_wmma_f32_16x16x32_f16(
              false, a, false, bb, (short)0, c, false, false);
    }
    const int ridx = rbase + mn;
    const float bias = bp[ridx];
#pragma unroll
    for (int r = 0; r < 8; ++r) {
      int t = r + 8 * half;                    // C layout: VGPR r -> row r/r+8
      float v   = c[r] + bias;
      float ang = 6.283185307179586f *
                  __builtin_amdgcn_rcpf(1.0f + __expf(-v));   // 2*pi*sigmoid
      csel[ridx * kT + t] = __cosf(0.5f * ang);
      ssel[ridx * kT + t] = __sinf(0.5f * ang);
    }
  }
  __syncthreads();                              // pool f16 staging now dead

  // ---- build U_prep (16x16 unitary on ancilla space) in the pool ----
  if (tid < 256) {
    Ure[tid] = ((tid >> 4) == (tid & 15)) ? 1.0f : 0.0f;
    Uim[tid] = 0.0f;
  }
  __syncthreads();
  for (int ly = 0; ly < 4; ++ly) {
    for (int qi = 0; qi < 4; ++qi) {
      int pi = (ly * 4 + qi) * 2;
      int a = 1 << (3 - qi);
      int mask = a - 1;
      if (tid < 128) {                          // RY on ancilla qubit qi
        int col = tid >> 3, p = tid & 7;
        int r0 = ((p & ~mask) << 1) | (p & mask);
        int r1 = r0 + a;
        int i0 = r0 * 16 + col, i1 = r1 * 16 + col;
        float c = pcp[pi], s = psp[pi];
        float ar = Ure[i0], ai = Uim[i0], br = Ure[i1], bi = Uim[i1];
        Ure[i0] = c * ar - s * br;  Uim[i0] = c * ai - s * bi;
        Ure[i1] = s * ar + c * br;  Uim[i1] = s * ai + c * bi;
      }
      __syncthreads();
      if (tid < 128) {                          // RZ on ancilla qubit qi
        int col = tid >> 3, p = tid & 7;
        int r0 = ((p & ~mask) << 1) | (p & mask);
        int r1 = r0 + a;
        int i0 = r0 * 16 + col, i1 = r1 * 16 + col;
        float c = pcp[pi + 1], s = psp[pi + 1];
        float ar = Ure[i0], ai = Uim[i0], br = Ure[i1], bi = Uim[i1];
        Ure[i0] = c * ar + s * ai;  Uim[i0] = c * ai - s * ar;
        Ure[i1] = c * br - s * bi;  Uim[i1] = c * bi + s * br;
      }
      __syncthreads();
    }
    for (int i = 0; i < 3; ++i) {               // CNOT(anc i -> anc i+1)
      int ca = 1 << (3 - i), ta = 1 << (3 - (i + 1));
      if (tid < 64) {
        int col = tid >> 2, p = tid & 3;
        int lm = ta - 1, hm = ca - 1;           // ta < ca always
        int t0 = ((p & ~lm) << 1) | (p & lm);
        int r  = ((t0 & ~hm) << 1) | (t0 & hm);
        int r0 = r | ca, r1 = r0 | ta;
        int i0 = r0 * 16 + col, i1 = r1 * 16 + col;
        float tr = Ure[i0], ti = Uim[i0];
        Ure[i0] = Ure[i1];  Uim[i0] = Uim[i1];
        Ure[i1] = tr;       Uim[i1] = ti;
      }
      __syncthreads();
    }
  }

  // ---- contract: A0 = U*D0,  A1..3 = U*D_k*U^dag,  A4 = D4*U^dag ----
  // stored transposed for applyA: A*[m*256 + t*16 + j] = A_m[j][t]
  for (int e = tid; e < 5 * 256; e += kThr) {
    int m = e >> 8, idx = e & 255;
    int t = idx >> 4, j = idx & 15;
    float vr, vi;
    if (m == 0) {
      float dc = sgc[0], ds = (t == 0) ? sgs[0] : -sgs[0];
      float ar = Ure[j * 16 + t], ai = Uim[j * 16 + t];
      vr = ar * dc - ai * ds;  vi = ar * ds + ai * dc;
    } else if (m == 4) {
      float dc = sgc[4], ds = (j == 0) ? sgs[4] : -sgs[4];
      float br = Ure[t * 16 + j], bi = Uim[t * 16 + j];
      vr = dc * br + ds * bi;  vi = ds * br - dc * bi;
    } else {
      float dc = sgc[m];
      float ar_ = 0.0f, ai_ = 0.0f;
#pragma unroll
      for (int u = 0; u < 16; ++u) {
        float ds = (u == 0) ? sgs[m] : -sgs[m];
        float xr = Ure[j * 16 + u], xi = Uim[j * 16 + u];
        float pr = xr * dc - xi * ds, pq = xr * ds + xi * dc;
        float br = Ure[t * 16 + u], bi = Uim[t * 16 + u];
        ar_ += pr * br + pq * bi;                // p * conj(b)
        ai_ += pq * br - pr * bi;
      }
      vr = ar_;  vi = ai_;
    }
    Are[m * 256 + idx] = vr;
    Aim[m * 256 + idx] = vi;
  }
  __syncthreads();

  // ---- circuit: A0, (S_k, A_{k+1}) x4, qff ----
  applyA(sre, sim_, Are, Aim, tid);
  __syncthreads();
  for (int k = 0; k < 4; ++k) {
    run_select(sre, sim_, csel, ssel, /*adj=*/(k & 1) != 0, tid);
    applyA(sre, sim_, Are + (k + 1) * 256, Aim + (k + 1) * 256, tid);
    __syncthreads();
  }

  // qff sim14: 1 layer, scalar angles, qubits 0..9
  for (int pr = 0; pr < 5; ++pr) {
    int g0 = 2 * pr;
    g_ryry(sre, sim_, strideOf(2 * pr), strideOf(2 * pr + 1),
           qcp[g0], qsp[g0], qcp[g0 + 1], qsp[g0 + 1], tid);
    __syncthreads();
  }
  for (int k = 0; k < 10; ++k) {
    int g = 10 + k, i = 9 - k;
    g_crx(sre, sim_, strideOf(i), strideOf((i + 1) % 10), qcp[g], qsp[g], tid);
    __syncthreads();
  }
  for (int pr = 0; pr < 5; ++pr) {
    int g0 = 20 + 2 * pr;
    g_ryry(sre, sim_, strideOf(2 * pr), strideOf(2 * pr + 1),
           qcp[g0], qsp[g0], qcp[g0 + 1], qsp[g0 + 1], tid);
    __syncthreads();
  }
  for (int k = 0; k < 10; ++k) {
    int g = 30 + k, i = (k == 0) ? 9 : (k - 1);
    g_crx(sre, sim_, strideOf(i), strideOf((i + 9) % 10), qcp[g], qsp[g], tid);
    __syncthreads();
  }

  // ---- expectation values X/Y/Z per data qubit ----
  float ev[30];
#pragma unroll
  for (int k = 0; k < 30; ++k) ev[k] = 0.0f;
  for (int q = 0; q < kNQ; ++q) {
    const int st = strideOf(q);
    const int mask = st - 1;
    float xr = 0.0f, xi = 0.0f, zz = 0.0f;
    for (int p = tid; p < kPairs; p += kThr) {
      int i0 = ((p & ~mask) << 1) | (p & mask);
      int i1 = i0 + st;
      float r0 = sre[i0], m0 = sim_[i0], r1 = sre[i1], m1 = sim_[i1];
      xr += r0 * r1 + m0 * m1;         // Re(conj(a0)*a1)
      xi += r0 * m1 - m0 * r1;         // Im(conj(a0)*a1)
      zz += r0 * r0 + m0 * m0 - r1 * r1 - m1 * m1;
    }
    ev[q]      = 2.0f * xr;
    ev[10 + q] = 2.0f * xi;
    ev[20 + q] = zz;
  }

  // deterministic reduction: shfl within wave, LDS across waves
  float o = 0.0f;
  for (int k = 0; k < 30; ++k) {
    float v = ev[k];
#pragma unroll
    for (int off = 16; off > 0; off >>= 1) v += __shfl_down(v, off, 32);
    if (lane == 0) wred[wave] = v;
    __syncthreads();
    if (wave == 0) {
      float u = wred[lane];
#pragma unroll
      for (int off = 16; off > 0; off >>= 1) u += __shfl_down(u, off, 32);
      if (lane == 0) o += u * Wout[k];
    }
    __syncthreads();
  }
  if (tid == 0) out[b] = o + bout[0];
}

}  // namespace

extern "C" void kernel_launch(void* const* d_in, const int* in_sizes, int n_in,
                              void* d_out, int out_size, void* d_ws, size_t ws_size,
                              hipStream_t stream) {
  const float* x    = (const float*)d_in[0];
  const float* Wp   = (const float*)d_in[1];
  const float* bp   = (const float*)d_in[2];
  const float* prep = (const float*)d_in[3];
  const float* sig  = (const float*)d_in[4];
  const float* qff  = (const float*)d_in[5];
  const float* Wout = (const float*)d_in[6];
  const float* bout = (const float*)d_in[7];
  float* out = (float*)d_out;

  const int B = in_sizes[0] / (kF * kT);   // 512
  qts_fused_kernel<<<dim3(B), dim3(kThr), 0, stream>>>(
      x, Wp, bp, prep, sig, qff, Wout, bout, out);
}
